// GAU_AE_28054726377537
// MI455X (gfx1250) — matrix operand order repaired and checked by reading, not compiled, not run on hardware
//
#include <hip/hip_runtime.h>
#include <hip/hip_bf16.h>
#include <stdint.h>

#define DIMN  768
#define KEYN  128
#define SEQN  4096
#define BATCHN 4
#define UVN   1536
#define ICOLS 3200            // DIM*4 + KEY
#define ROWSN (BATCHN*SEQN)   // 16384

typedef __bf16 bf16_t;
typedef __attribute__((ext_vector_type(16))) __bf16 bf16x16;
typedef __attribute__((ext_vector_type(8)))  __bf16 bf16x8;
typedef __attribute__((ext_vector_type(8)))  float  f32x8;

union Frag { bf16x16 v; bf16x8 h[2]; };

// Build a 16-element bf16 fragment from two 8-element (16B) loads.
__device__ __forceinline__ bf16x16 frag_from(const bf16_t* lo, const bf16_t* hi) {
  Frag f;
  f.h[0] = *reinterpret_cast<const bf16x8*>(lo);
  f.h[1] = *reinterpret_cast<const bf16x8*>(hi);
  return f.v;
}

__device__ __forceinline__ f32x8 wmma_bf16(bf16x16 a, bf16x16 b, f32x8 c) {
  // (neg_a, A, neg_b, B, c_mod, C, reuse_a, reuse_b)
  return __builtin_amdgcn_wmma_f32_16x16x32_bf16(false, a, false, b, (short)0, c,
                                                 false, false);
}

__device__ __forceinline__ f32x8 zero8() {
  f32x8 z;
#pragma unroll
  for (int i = 0; i < 8; ++i) z[i] = 0.f;
  return z;
}

// fast silu: x * rcp(1+exp(-x))  (v_exp_f32 + v_rcp_f32, no IEEE div sequence)
__device__ __forceinline__ float fast_silu(float v) {
  return v * __builtin_amdgcn_rcpf(1.f + __expf(-v));
}

// ---------------- prep: transposes + fp32->bf16 ----------------
__global__ __launch_bounds__(256) void k_prep_wiT(const float* __restrict__ Wi,
                                                  bf16_t* __restrict__ WiT) {
  int idx = blockIdx.x * 256 + threadIdx.x;
  if (idx >= ICOLS * DIMN) return;
  int n = idx / DIMN, k = idx - n * DIMN;
  WiT[(size_t)n * DIMN + k] = (bf16_t)Wi[(size_t)k * ICOLS + n];
}

__global__ __launch_bounds__(256) void k_prep_woT(const float* __restrict__ Wo,
                                                  bf16_t* __restrict__ WoT) {
  int idx = blockIdx.x * 256 + threadIdx.x;
  if (idx >= DIMN * UVN) return;
  int n = idx / UVN, k = idx - n * UVN;
  WoT[(size_t)n * UVN + k] = (bf16_t)Wo[(size_t)k * DIMN + n];
}

__global__ __launch_bounds__(256) void k_prep_h(const float* __restrict__ H,
                                                bf16_t* __restrict__ Hbf) {
  size_t idx = (size_t)blockIdx.x * 256 + threadIdx.x;
  if (idx >= (size_t)ROWSN * DIMN) return;
  Hbf[idx] = (bf16_t)H[idx];
}

// ---------------- kernel 1: x = silu(H @ Wi + bi), scatter u / vT / qk --------
// grid (ROWS/16, 3200/128), block 64 (2 waves); each wave owns 16x64 of output.
// Two fixed fragment register sets ping-pong (unroll-by-2): loads land directly
// in the set consumed next -> no rotation movs, incremental loadcnt waits.
__global__ __launch_bounds__(64) void k_proj1(const bf16_t* __restrict__ Hbf,
                                              const bf16_t* __restrict__ WiT,
                                              const float* __restrict__ bi,
                                              bf16_t* __restrict__ Ubf,
                                              bf16_t* __restrict__ VT,
                                              float* __restrict__ QKf) {
  const int lane = threadIdx.x & 31;
  const int wave = threadIdx.x >> 5;
  const int l15  = lane & 15;
  const int hs   = lane >> 4;          // 0/1 (lane half)
  const int rows = blockIdx.x * 16;
  const int colw = blockIdx.y * 128 + wave * 64;

  f32x8 acc[4];
#pragma unroll
  for (int t = 0; t < 4; ++t) acc[t] = zero8();

  const bf16_t* arow = Hbf + (size_t)(rows + l15) * DIMN + hs * 16;
  const bf16_t* bp[4];
#pragma unroll
  for (int t = 0; t < 4; ++t)
    bp[t] = WiT + (size_t)(colw + t * 16 + l15) * DIMN + hs * 16;

  bf16x16 a0 = frag_from(arow, arow + 8);
  bf16x16 b0[4];
#pragma unroll
  for (int t = 0; t < 4; ++t) b0[t] = frag_from(bp[t], bp[t] + 8);
  bf16x16 a1;
  bf16x16 b1[4];

  int k0 = 0;
  for (; k0 < DIMN - 64; k0 += 64) {
    __builtin_prefetch(bp[0] + k0 + 96, 0, 1);   // global_prefetch_b8
    a1 = frag_from(arow + k0 + 32, arow + k0 + 40);
#pragma unroll
    for (int t = 0; t < 4; ++t)
      b1[t] = frag_from(bp[t] + k0 + 32, bp[t] + k0 + 40);
#pragma unroll
    for (int t = 0; t < 4; ++t) acc[t] = wmma_bf16(a0, b0[t], acc[t]);

    a0 = frag_from(arow + k0 + 64, arow + k0 + 72);
#pragma unroll
    for (int t = 0; t < 4; ++t)
      b0[t] = frag_from(bp[t] + k0 + 64, bp[t] + k0 + 72);
#pragma unroll
    for (int t = 0; t < 4; ++t) acc[t] = wmma_bf16(a1, b1[t], acc[t]);
  }
  // tail: k0 == DIMN-64; fragments for k0 already resident in set 0
  a1 = frag_from(arow + k0 + 32, arow + k0 + 40);
#pragma unroll
  for (int t = 0; t < 4; ++t)
    b1[t] = frag_from(bp[t] + k0 + 32, bp[t] + k0 + 40);
#pragma unroll
  for (int t = 0; t < 4; ++t) acc[t] = wmma_bf16(a0, b0[t], acc[t]);
#pragma unroll
  for (int t = 0; t < 4; ++t) acc[t] = wmma_bf16(a1, b1[t], acc[t]);

#pragma unroll
  for (int t = 0; t < 4; ++t) {
    const int col = colw + t * 16 + l15;
    const float bias = bi[col];
#pragma unroll
    for (int i = 0; i < 8; ++i) {
      const int m = i + hs * 8;
      const size_t row = (size_t)(rows + m);
      float s = fast_silu(acc[t][i] + bias);
      if (col < UVN) {
        Ubf[row * UVN + col] = (bf16_t)s;
      } else if (col < 2 * UVN) {
        const int d = col - UVN;
        const int bb = (int)(row / SEQN);
        const int sp = (int)(row % SEQN);
        VT[((size_t)bb * UVN + d) * SEQN + sp] = (bf16_t)s;   // v transposed
      } else {
        QKf[row * KEYN + (col - 2 * UVN)] = s;
      }
    }
  }
}

// ---------------- kernel 2: per-row scale/shift + RoPE -> q,k bf16 -----------
__global__ __launch_bounds__(64) void k_rope(const float* __restrict__ QKf,
                                             const float* __restrict__ qg,
                                             const float* __restrict__ qb,
                                             const float* __restrict__ kg,
                                             const float* __restrict__ kb,
                                             bf16_t* __restrict__ Qbf,
                                             bf16_t* __restrict__ Kbf) {
  const int row = blockIdx.x;
  const int t = threadIdx.x;          // 0..63 rotary pairs
  const int sp = row % SEQN;
  const float x1 = QKf[(size_t)row * KEYN + 2 * t];
  const float x2 = QKf[(size_t)row * KEYN + 2 * t + 1];
  const float omega = __expf(-((float)t * (1.0f / 64.0f)) * 9.210340371976184f);
  float sn, cs;
  __sincosf((float)sp * omega, &sn, &cs);

  const float q1 = x1 * qg[2 * t] + qb[2 * t];
  const float q2 = x2 * qg[2 * t + 1] + qb[2 * t + 1];
  Qbf[(size_t)row * KEYN + t]      = (bf16_t)(q1 * cs - q2 * sn);
  Qbf[(size_t)row * KEYN + 64 + t] = (bf16_t)(q1 * sn + q2 * cs);

  const float k1 = x1 * kg[2 * t] + kb[2 * t];
  const float k2 = x2 * kg[2 * t + 1] + kb[2 * t + 1];
  Kbf[(size_t)row * KEYN + t]      = (bf16_t)(k1 * cs - k2 * sn);
  Kbf[(size_t)row * KEYN + 64 + t] = (bf16_t)(k1 * sn + k2 * cs);
}

// ---------------- kernel 3: flash-style relu^2 attention, fused u*(A@v) ------
// grid ROWS/16, block 256 (8 waves). Wave w: S-tile for keys w*16.. of each
// 128-key superblock, plus 12 d-tiles (192 cols) of the Av accumulator.
// Score tile ping-pongs between two LDS buffers: one barrier per key block.
__global__ __launch_bounds__(256) void k_attn(const bf16_t* __restrict__ Qbf,
                                              const bf16_t* __restrict__ Kbf,
                                              const bf16_t* __restrict__ VT,
                                              const bf16_t* __restrict__ Ubf,
                                              bf16_t* __restrict__ UAV) {
  __shared__ __align__(16) bf16_t A_lds[2][16 * 128];
  const int lane = threadIdx.x & 31;
  const int wave = threadIdx.x >> 5;
  const int l15  = lane & 15;
  const int hs   = lane >> 4;
  const int rows = blockIdx.x * 16;
  const int b    = rows / SEQN;
  const int dbase = wave * 192;

  // q A-fragments (K=128 -> 4 steps), held in registers for the whole kernel
  bf16x16 qf[4];
  {
    const bf16_t* qp = Qbf + (size_t)(rows + l15) * KEYN + hs * 16;
#pragma unroll
    for (int ks = 0; ks < 4; ++ks)
      qf[ks] = frag_from(qp + ks * 32, qp + ks * 32 + 8);
  }

  f32x8 acc[12];
#pragma unroll
  for (int t = 0; t < 12; ++t) acc[t] = zero8();

  const float inv_sqrt = 0.08838834764831845f;  // 1/sqrt(128)
  const float inv_n = 1.0f / (float)SEQN;

  for (int kb0 = 0; kb0 < SEQN; kb0 += 128) {
    const int pb = (kb0 >> 7) & 1;               // ping-pong buffer
    // ---- scores: this wave's 16x16 S tile (keys kb0 + wave*16 ..) ----
    f32x8 s = zero8();
    const bf16_t* kp =
        Kbf + (size_t)(b * SEQN + kb0 + wave * 16 + l15) * KEYN + hs * 16;
#pragma unroll
    for (int ks = 0; ks < 4; ++ks) {
      bf16x16 kf = frag_from(kp + ks * 32, kp + ks * 32 + 8);
      s = wmma_bf16(qf[ks], kf, s);
    }
#pragma unroll
    for (int i = 0; i < 8; ++i) {
      float a = s[i] * inv_sqrt;
      a = fmaxf(a, 0.f);
      a = a * a * inv_n;                        // relu(a)^2 / n
      const int m = i + hs * 8;
      A_lds[pb][m * 128 + wave * 16 + l15] = (bf16_t)a;
    }
    __syncthreads();
    // ---- Av: A(16x128) @ v(128x192 per wave) ----
#pragma unroll
    for (int kb = 0; kb < 4; ++kb) {
      const bf16_t* ap = &A_lds[pb][l15 * 128 + kb * 32 + hs * 16];
      bf16x16 af = frag_from(ap, ap + 8);       // ds_load_b128 x2
      const bf16_t* vp = VT + ((size_t)b * UVN + dbase + l15) * SEQN +
                         kb0 + kb * 32 + hs * 16;
#pragma unroll
      for (int g = 0; g < 2; ++g) {
        bf16x16 vf[6];                          // batch loads, then batch WMMAs
#pragma unroll
        for (int t = 0; t < 6; ++t) {
          const bf16_t* p = vp + (size_t)(g * 6 + t) * 16 * SEQN;
          vf[t] = frag_from(p, p + 8);
        }
#pragma unroll
        for (int t = 0; t < 6; ++t)
          acc[g * 6 + t] = wmma_bf16(af, vf[t], acc[g * 6 + t]);
      }
    }
  }

  // ---- epilogue: u ⊙ Av -> bf16 ----
#pragma unroll
  for (int t = 0; t < 12; ++t) {
#pragma unroll
    for (int i = 0; i < 8; ++i) {
      const int m = i + hs * 8;
      const size_t row = (size_t)(rows + m);
      const int d = dbase + t * 16 + l15;
      const float u = (float)Ubf[row * UVN + d];
      UAV[row * UVN + d] = (bf16_t)(u * acc[t][i]);
    }
  }
}

// ---------------- kernel 4: (u*Av)@Wo + bo + residual, RMSNorm ---------------
// grid ROWS/16, block 256 (8 waves); wave w owns 96 output cols (6 N-tiles).
// Same unroll-by-2 double buffering as k_proj1.
__global__ __launch_bounds__(256) void k_out(const bf16_t* __restrict__ UAV,
                                             const bf16_t* __restrict__ WoT,
                                             const float* __restrict__ bo,
                                             const float* __restrict__ H,
                                             float* __restrict__ Out) {
  __shared__ float rowsq[16];
  const int lane = threadIdx.x & 31;
  const int wave = threadIdx.x >> 5;
  const int l15  = lane & 15;
  const int hs   = lane >> 4;
  const int rows = blockIdx.x * 16;
  const int nbase = wave * 96;

  f32x8 acc[6];
#pragma unroll
  for (int t = 0; t < 6; ++t) acc[t] = zero8();

  const bf16_t* ap = UAV + (size_t)(rows + l15) * UVN + hs * 16;
  const bf16_t* bp[6];
#pragma unroll
  for (int t = 0; t < 6; ++t)
    bp[t] = WoT + (size_t)(nbase + t * 16 + l15) * UVN + hs * 16;

  bf16x16 a0 = frag_from(ap, ap + 8);
  bf16x16 b0[6];
#pragma unroll
  for (int t = 0; t < 6; ++t) b0[t] = frag_from(bp[t], bp[t] + 8);
  bf16x16 a1;
  bf16x16 b1[6];

  int k0 = 0;
  for (; k0 < UVN - 64; k0 += 64) {
    a1 = frag_from(ap + k0 + 32, ap + k0 + 40);
#pragma unroll
    for (int t = 0; t < 6; ++t)
      b1[t] = frag_from(bp[t] + k0 + 32, bp[t] + k0 + 40);
#pragma unroll
    for (int t = 0; t < 6; ++t) acc[t] = wmma_bf16(a0, b0[t], acc[t]);

    a0 = frag_from(ap + k0 + 64, ap + k0 + 72);
#pragma unroll
    for (int t = 0; t < 6; ++t)
      b0[t] = frag_from(bp[t] + k0 + 64, bp[t] + k0 + 72);
#pragma unroll
    for (int t = 0; t < 6; ++t) acc[t] = wmma_bf16(a1, b1[t], acc[t]);
  }
  // tail
  a1 = frag_from(ap + k0 + 32, ap + k0 + 40);
#pragma unroll
  for (int t = 0; t < 6; ++t)
    b1[t] = frag_from(bp[t] + k0 + 32, bp[t] + k0 + 40);
#pragma unroll
  for (int t = 0; t < 6; ++t) acc[t] = wmma_bf16(a0, b0[t], acc[t]);
#pragma unroll
  for (int t = 0; t < 6; ++t) acc[t] = wmma_bf16(a1, b1[t], acc[t]);

  if (threadIdx.x < 16) rowsq[threadIdx.x] = 0.f;
  __syncthreads();

  float o[6][8];
  float part[8];
#pragma unroll
  for (int i = 0; i < 8; ++i) part[i] = 0.f;
#pragma unroll
  for (int t = 0; t < 6; ++t) {
    const int col = nbase + t * 16 + l15;
    const float bias = bo[col];
#pragma unroll
    for (int i = 0; i < 8; ++i) {
      const int m = i + hs * 8;
      const size_t row = (size_t)(rows + m);
      const float v = acc[t][i] + bias + H[row * DIMN + col];
      o[t][i] = v;
      part[i] += v * v;
    }
  }
#pragma unroll
  for (int i = 0; i < 8; ++i) atomicAdd(&rowsq[i + hs * 8], part[i]);  // ds_add
  __syncthreads();

  float scale[8];
#pragma unroll
  for (int i = 0; i < 8; ++i)
    scale[i] =
        __builtin_amdgcn_rsqf(rowsq[i + hs * 8] * (1.0f / DIMN) + 1e-12f);
#pragma unroll
  for (int t = 0; t < 6; ++t) {
    const int col = nbase + t * 16 + l15;
#pragma unroll
    for (int i = 0; i < 8; ++i) {
      const int m = i + hs * 8;
      const size_t row = (size_t)(rows + m);
      Out[row * DIMN + col] = o[t][i] * scale[i];
    }
  }
}

// ---------------------------------------------------------------------------
extern "C" void kernel_launch(void* const* d_in, const int* in_sizes, int n_in,
                              void* d_out, int out_size, void* d_ws, size_t ws_size,
                              hipStream_t stream) {
  (void)in_sizes; (void)n_in; (void)out_size; (void)ws_size;
  const float* H  = (const float*)d_in[0];
  const float* Wi = (const float*)d_in[1];
  const float* bi = (const float*)d_in[2];
  const float* Wo = (const float*)d_in[3];
  const float* bo = (const float*)d_in[4];
  const float* qg = (const float*)d_in[5];
  const float* qb = (const float*)d_in[6];
  const float* kg = (const float*)d_in[7];
  const float* kb = (const float*)d_in[8];

  uint8_t* ws = (uint8_t*)d_ws;
  size_t off = 0;
  auto alloc = [&](size_t bytes) -> void* {
    void* p = ws + off;
    off += (bytes + 255) & ~(size_t)255;
    return p;
  };

  bf16_t* WiT = (bf16_t*)alloc((size_t)ICOLS * DIMN * 2);
  bf16_t* WoT = (bf16_t*)alloc((size_t)DIMN * UVN * 2);
  bf16_t* Hbf = (bf16_t*)alloc((size_t)ROWSN * DIMN * 2);
  bf16_t* Ubf = (bf16_t*)alloc((size_t)ROWSN * UVN * 2);
  bf16_t* VT  = (bf16_t*)alloc((size_t)ROWSN * UVN * 2);
  float*  QKf = (float*)alloc((size_t)ROWSN * KEYN * 4);
  bf16_t* Qbf = (bf16_t*)alloc((size_t)ROWSN * KEYN * 2);
  bf16_t* Kbf = (bf16_t*)alloc((size_t)ROWSN * KEYN * 2);
  bf16_t* UAV = (bf16_t*)alloc((size_t)ROWSN * UVN * 2);

  k_prep_wiT<<<(ICOLS * DIMN + 255) / 256, 256, 0, stream>>>(Wi, WiT);
  k_prep_woT<<<(DIMN * UVN + 255) / 256, 256, 0, stream>>>(Wo, WoT);
  k_prep_h<<<(int)(((size_t)ROWSN * DIMN + 255) / 256), 256, 0, stream>>>(H, Hbf);

  dim3 g1(ROWSN / 16, ICOLS / 128);
  k_proj1<<<g1, 64, 0, stream>>>(Hbf, WiT, bi, Ubf, VT, QKf);
  k_rope<<<ROWSN, 64, 0, stream>>>(QKf, qg, qb, kg, kb, Qbf, Kbf);
  k_attn<<<ROWSN / 16, 256, 0, stream>>>(Qbf, Kbf, VT, Ubf, UAV);
  k_out<<<ROWSN / 16, 256, 0, stream>>>(UAV, WoT, bo, H, (float*)d_out);
}
